// TitansLayer_62792421868020
// MI455X (gfx1250) — matrix operand order repaired and checked by reading, not compiled, not run on hardware
//
#include <hip/hip_runtime.h>
#include <cmath>

typedef __bf16 bf16_t;
typedef __attribute__((ext_vector_type(16))) __bf16 v16bf;
typedef __attribute__((ext_vector_type(8)))  float  v8f;
typedef __attribute__((ext_vector_type(4))) unsigned int u32x4;
typedef __attribute__((ext_vector_type(8))) int i32x8;
typedef __attribute__((ext_vector_type(4))) int i32x4;

#define LRATE  0.01f
#define WDECAY 0.01f

// ---- Tensor Data Mover availability / arity detection ----
#if defined(__HIP_DEVICE_COMPILE__) && __has_builtin(__builtin_amdgcn_tensor_load_to_lds)
#define USE_TDM 1
#else
#define USE_TDM 0
#endif
#if __has_include(<hip/amd_detail/amd_gfx1250_TDM.h>)
#define TDM_6ARG 1
#else
#define TDM_6ARG 0
#endif

static __device__ __forceinline__ float  bf2f(bf16_t x) { return (float)x; }
static __device__ __forceinline__ bf16_t f2bf(float x)  { return (bf16_t)x; }

// Load a 16-element bf16 WMMA fragment from two contiguous 16-byte chunks.
static __device__ __forceinline__ v16bf ld_frag(const bf16_t* p0, const bf16_t* p1) {
  union { uint4 u[2]; v16bf v; } f;
  f.u[0] = *(const uint4*)p0;
  f.u[1] = *(const uint4*)p1;
  return f.v;
}

#if USE_TDM
// Issue a TDM 2D tile load (bf16 elements) global -> LDS, with optional LDS row
// padding done by the TDM itself (pad codes per ISA 8.4: interval 0=2DW..7=256DW,
// amount 0=1DW..127=128DW). tensor_h = remaining valid rows (rows beyond it are
// zero-filled by TDM OOB handling). One call per WAVE (EXEC ignored by TDM).
static __device__ __forceinline__ void tdm_load_2d(
    unsigned lds_off, const void* gptr,
    unsigned tensor_w, unsigned tensor_h,
    unsigned tile_w, unsigned tile_h,
    unsigned row_stride, unsigned pad_int_code, unsigned pad_amt_code)
{
  unsigned long long ga = (unsigned long long)(size_t)gptr;
  u32x4 g0;
  g0[0] = 1u;                                               // count=1, user mode
  g0[1] = lds_off;                                          // LDS byte address
  g0[2] = (unsigned)(ga & 0xffffffffu);                     // global_addr[31:0]
  g0[3] = (unsigned)((ga >> 32) & 0x1ffffffu) | (2u << 30); // addr[56:32] | type=2
  i32x8 g1;
  g1[0] = (int)((1u << 16) | (1u << 20) |                   // data_size=2B, pad_en
                (pad_int_code << 22) | (pad_amt_code << 25));
  g1[1] = (int)(tensor_w << 16);                            // tensor_dim0[15:0]
  g1[2] = (int)((tensor_w >> 16) | (tensor_h << 16));       // dim0[31:16]|dim1[15:0]
  g1[3] = (int)((tensor_h >> 16) | (tile_w << 16));         // dim1[31:16]|tile_dim0
  g1[4] = (int)(tile_h & 0xffffu);                          // tile_dim1 (tile_dim2=0)
  g1[5] = (int)row_stride;                                  // tensor_dim0_stride[31:0]
  g1[6] = 0;
  g1[7] = 0;
  i32x4 z4 = {0, 0, 0, 0};
#if TDM_6ARG
  i32x8 z8 = {0, 0, 0, 0, 0, 0, 0, 0};
  __builtin_amdgcn_tensor_load_to_lds(g0, g1, z4, z4, z8, 0);
#else
  __builtin_amdgcn_tensor_load_to_lds(g0, g1, z4, z4, 0);
#endif
}
#endif

// ---------------------------------------------------------------------------
// Generic bf16 WMMA GEMM:  C[M,N] = X[M,K] @ W[N,K]^T (+bias)
// (M%64==0, N%64==0 or N==64, K%32==0). Tile 64x64, BK=32, 256 threads:
// 8 waves = 4 row-blocks x 2 col-groups; each wave holds 2 accumulators and
// issues 2 WMMAs per K-step from 1 A-frag + 2 B-frags. Tiles arrive via TDM
// (hardware-padded LDS rows, stride 40 bf16) or coop b128 loads as fallback.
// Epilogues:
//  0: f32out = v                      1: b16out = v
//  2: f32out = v; b16out = silu(v)    3: b16out = scl*(v - aux)        (dpred)
//  4: b16out = v * silu'(aux)         5: b16out = silu(v)
//  6: o = v + f32out; f32out = o; b16out = o                          (attn += mem)
// ---------------------------------------------------------------------------
template<int EPI>
__global__ __launch_bounds__(256) void gemm_bf16_k(
    const bf16_t* __restrict__ X, const bf16_t* __restrict__ Wm,
    const float* __restrict__ bias, int M, int N, int K,
    float* __restrict__ f32out, bf16_t* __restrict__ b16out,
    const float* __restrict__ aux, float scl)
{
  __shared__ bf16_t As[64 * 40];
  __shared__ bf16_t Ws[64 * 40];
  const int tid  = threadIdx.x;
  const int wave = tid >> 5, l = tid & 31, half = l >> 4, nn = l & 15;
  const int rowblk = wave & 3, colg = wave >> 2;
  const int row0 = blockIdx.x * 64, col0 = blockIdx.y * 64;

  const v8f zf = {0.f,0.f,0.f,0.f,0.f,0.f,0.f,0.f};
  v8f acc0 = zf, acc1 = zf;

  for (int k0 = 0; k0 < K; k0 += 32) {
    __syncthreads();
#if USE_TDM
    if (wave == 0) {
      tdm_load_2d((unsigned)(size_t)&As[0], X + (size_t)row0 * K + k0,
                  (unsigned)(K - k0), (unsigned)(M - row0), 32, 64,
                  (unsigned)K, 3, 3);                 // 64B rows, 16B pad -> stride 40
      tdm_load_2d((unsigned)(size_t)&Ws[0], Wm + (size_t)col0 * K + k0,
                  (unsigned)(K - k0), (unsigned)(N - col0), 32, 64,
                  (unsigned)K, 3, 3);
      __builtin_amdgcn_s_wait_tensorcnt(0);
    }
#else
    {
      const int ar = tid >> 2, ac = (tid & 3) * 8;   // 64 rows x 4 chunks of 8 bf16
      *(uint4*)&As[ar * 40 + ac] =
          *(const uint4*)&X[(size_t)(row0 + ar) * K + k0 + ac];
      *(uint4*)&Ws[ar * 40 + ac] =
          *(const uint4*)&Wm[(size_t)(col0 + ar) * K + k0 + ac];
    }
#endif
    __syncthreads();
    v16bf af  = ld_frag(&As[(rowblk * 16 + nn) * 40 + half * 8],
                        &As[(rowblk * 16 + nn) * 40 + 16 + half * 8]);
    v16bf bf0 = ld_frag(&Ws[(colg * 32 + nn) * 40 + half * 8],
                        &Ws[(colg * 32 + nn) * 40 + 16 + half * 8]);
    v16bf bf1 = ld_frag(&Ws[(colg * 32 + 16 + nn) * 40 + half * 8],
                        &Ws[(colg * 32 + 16 + nn) * 40 + 16 + half * 8]);
    acc0 = __builtin_amdgcn_wmma_f32_16x16x32_bf16(false, af, false, bf0,
                                                   (short)0, acc0, false, false);
    acc1 = __builtin_amdgcn_wmma_f32_16x16x32_bf16(false, af, false, bf1,
                                                   (short)0, acc1, false, false);
  }

  const int gm0 = row0 + rowblk * 16 + half * 8;
  auto epi = [&](const v8f& a, int gn) {
    const float bv = bias ? bias[gn] : 0.f;
#pragma unroll
    for (int r = 0; r < 8; r++) {
      const size_t idx = (size_t)(gm0 + r) * N + gn;
      float v = a[r] + bv;
      if constexpr (EPI == 0) { f32out[idx] = v; }
      else if constexpr (EPI == 1) { b16out[idx] = f2bf(v); }
      else if constexpr (EPI == 2) {
        f32out[idx] = v; float s = 1.f / (1.f + __expf(-v)); b16out[idx] = f2bf(v * s);
      }
      else if constexpr (EPI == 3) { b16out[idx] = f2bf(scl * (v - aux[idx])); }
      else if constexpr (EPI == 4) {
        float z = aux[idx]; float s = 1.f / (1.f + __expf(-z));
        float ds = s * (1.f + z * (1.f - s)); b16out[idx] = f2bf(v * ds);
      }
      else if constexpr (EPI == 5) {
        float s = 1.f / (1.f + __expf(-v)); b16out[idx] = f2bf(v * s);
      }
      else if constexpr (EPI == 6) {
        float o = v + f32out[idx]; f32out[idx] = o; b16out[idx] = f2bf(o);
      }
    }
  };
  epi(acc0, col0 + colg * 32 + nn);
  epi(acc1, col0 + colg * 32 + 16 + nn);
}

// ---------------------------------------------------------------------------
// Flash attention (WMMA, wave32). grid = (S/128, B*H), 256 threads.
// Each wave owns 16 query rows; 32-key tiles streamed through LDS. The K tile
// comes in via TDM (zero-fill past T via tensor_dim1); V is loaded transposed
// by the threads so PV B-frags are contiguous b128 LDS reads.
// ---------------------------------------------------------------------------
#define FA_S 2048
#define FA_T 2064
#define FA_TPAD 2080
#define FA_D 1024
#define FA_H 8

__global__ __launch_bounds__(256) void flash_attn_k(
    const bf16_t* __restrict__ qb, const bf16_t* __restrict__ kb,
    const bf16_t* __restrict__ vb, bf16_t* __restrict__ ctxb)
{
  __shared__ bf16_t Ks[32 * 136];       // keys row-major (32 x 128 +pad)
  __shared__ bf16_t Vst[128 * 40];      // V transposed (col x key +pad)
  __shared__ bf16_t Ps[8 * 16 * 40];    // per-wave P tile (16 x 32 +pad)

  const int tid  = threadIdx.x;
  const int wave = tid >> 5, l = tid & 31, half = l >> 4, nn = l & 15;
  const int b = blockIdx.y / FA_H, h = blockIdx.y % FA_H;
  const int qbase = blockIdx.x * 128;
  const float scl = 0.08838834764831845f;   // 1/sqrt(128)

  // Q fragments (16x128 per wave) kept in registers, A-layout.
  v16bf qf[4];
  {
    const bf16_t* qp = qb + ((size_t)(b * FA_S + qbase + wave * 16 + nn)) * FA_D + h * 128;
#pragma unroll
    for (int kk = 0; kk < 4; kk++)
      qf[kk] = ld_frag(qp + kk * 32 + half * 8, qp + kk * 32 + 16 + half * 8);
  }

  const v8f zf = {0.f,0.f,0.f,0.f,0.f,0.f,0.f,0.f};
  v8f o[8];
#pragma unroll
  for (int c = 0; c < 8; c++) o[c] = zf;
  float mrow[8], lrow[8];
#pragma unroll
  for (int j = 0; j < 8; j++) { mrow[j] = -1e30f; lrow[j] = 0.f; }

  const int krow = tid >> 3, kcol = (tid & 7) * 16;
  bf16_t* ps = Ps + wave * 16 * 40;

  for (int t0 = 0; t0 < FA_TPAD; t0 += 32) {
    // -- K tile via TDM (or coop loads); V tile transposed by threads --
#if USE_TDM
    if (wave == 0) {
      tdm_load_2d((unsigned)(size_t)&Ks[0],
                  kb + ((size_t)(b * FA_T + t0)) * FA_D + h * 128,
                  128u, (unsigned)(FA_T - t0), 128, 32,
                  (unsigned)FA_D, 5, 3);              // 256B rows, 16B pad -> stride 136
    }
#else
    {
      uint4 u0 = {0,0,0,0}, u1 = {0,0,0,0};
      if (t0 + krow < FA_T) {
        const uint4* src = (const uint4*)&kb[((size_t)(b * FA_T + t0 + krow)) * FA_D + h * 128 + kcol];
        u0 = src[0]; u1 = src[1];
      }
      *(uint4*)&Ks[krow * 136 + kcol]     = u0;
      *(uint4*)&Ks[krow * 136 + kcol + 8] = u1;
    }
#endif
    {
      union { uint4 u; bf16_t e[8]; } a0, a1;
      a0.u = (uint4){0,0,0,0}; a1.u = (uint4){0,0,0,0};
      if (t0 + krow < FA_T) {
        const uint4* src = (const uint4*)&vb[((size_t)(b * FA_T + t0 + krow)) * FA_D + h * 128 + kcol];
        a0.u = src[0]; a1.u = src[1];
      }
#pragma unroll
      for (int i = 0; i < 8; i++) Vst[(kcol + i) * 40 + krow] = a0.e[i];
#pragma unroll
      for (int i = 0; i < 8; i++) Vst[(kcol + 8 + i) * 40 + krow] = a1.e[i];
    }
#if USE_TDM
    if (wave == 0) __builtin_amdgcn_s_wait_tensorcnt(0);
#endif
    __syncthreads();

    // -- S = Q @ K^T (two 16x16 tiles) --
    v8f s0 = zf, s1 = zf;
#pragma unroll
    for (int kk = 0; kk < 4; kk++) {
      v16bf kf = ld_frag(&Ks[nn * 136 + kk * 32 + half * 8],
                         &Ks[nn * 136 + kk * 32 + 16 + half * 8]);
      s0 = __builtin_amdgcn_wmma_f32_16x16x32_bf16(false, qf[kk], false, kf,
                                                   (short)0, s0, false, false);
    }
#pragma unroll
    for (int kk = 0; kk < 4; kk++) {
      v16bf kf = ld_frag(&Ks[(16 + nn) * 136 + kk * 32 + half * 8],
                         &Ks[(16 + nn) * 136 + kk * 32 + 16 + half * 8]);
      s1 = __builtin_amdgcn_wmma_f32_16x16x32_bf16(false, qf[kk], false, kf,
                                                   (short)0, s1, false, false);
    }

    // -- online softmax --
    const bool msk0 = (t0 + nn) >= FA_T;
    const bool msk1 = (t0 + 16 + nn) >= FA_T;
    float p0[8], p1[8], corr[8];
#pragma unroll
    for (int j = 0; j < 8; j++) {
      float a0 = msk0 ? -1e30f : s0[j] * scl;
      float a1 = msk1 ? -1e30f : s1[j] * scl;
      float mx = fmaxf(a0, a1);
      mx = fmaxf(mx, __shfl_xor(mx, 1));
      mx = fmaxf(mx, __shfl_xor(mx, 2));
      mx = fmaxf(mx, __shfl_xor(mx, 4));
      mx = fmaxf(mx, __shfl_xor(mx, 8));
      float mnew = fmaxf(mrow[j], mx);
      float cr = __expf(mrow[j] - mnew);
      float e0 = __expf(a0 - mnew), e1 = __expf(a1 - mnew);
      float rs = e0 + e1;
      rs += __shfl_xor(rs, 1); rs += __shfl_xor(rs, 2);
      rs += __shfl_xor(rs, 4); rs += __shfl_xor(rs, 8);
      lrow[j] = lrow[j] * cr + rs;
      mrow[j] = mnew; corr[j] = cr; p0[j] = e0; p1[j] = e1;
    }
#pragma unroll
    for (int c = 0; c < 8; c++)
#pragma unroll
      for (int j = 0; j < 8; j++) o[c][j] *= corr[j];

    // write P tile (C-layout -> LDS row-major), then re-read in A-layout
#pragma unroll
    for (int j = 0; j < 8; j++) {
      int row = half * 8 + j;
      ps[row * 40 + nn]      = f2bf(p0[j]);
      ps[row * 40 + 16 + nn] = f2bf(p1[j]);
    }
    __syncthreads();

    // -- O += P @ V --
    v16bf pf = ld_frag(&ps[nn * 40 + half * 8], &ps[nn * 40 + 16 + half * 8]);
#pragma unroll
    for (int c = 0; c < 8; c++) {
      v16bf vf = ld_frag(&Vst[(c * 16 + nn) * 40 + half * 8],
                         &Vst[(c * 16 + nn) * 40 + 16 + half * 8]);
      o[c] = __builtin_amdgcn_wmma_f32_16x16x32_bf16(false, pf, false, vf,
                                                     (short)0, o[c], false, false);
    }
    __syncthreads();
  }

  // -- normalize + store ctx (bf16) --
#pragma unroll
  for (int c = 0; c < 8; c++)
#pragma unroll
    for (int j = 0; j < 8; j++) {
      int row = qbase + wave * 16 + half * 8 + j;
      float v = o[c][j] / lrow[j];
      ctxb[((size_t)(b * FA_S + row)) * FA_D + h * 128 + c * 16 + nn] = f2bf(v);
    }
}

// ---------------------------------------------------------------------------
// Small helper kernels
// ---------------------------------------------------------------------------
__global__ void f32_to_bf16_k(const float* __restrict__ s, bf16_t* __restrict__ d, int n) {
  int i = blockIdx.x * 256 + threadIdx.x;
  if (i < n) d[i] = f2bf(s[i]);
}

__global__ void build_kv_k(const float* __restrict__ x, const float* __restrict__ persist,
                           bf16_t* __restrict__ kv) {
  size_t i = (size_t)blockIdx.x * 256 + threadIdx.x;
  int d = (int)(i % 1024);
  size_t r = i / 1024;
  int b = (int)(r / 2064), t = (int)(r % 2064);
  float v = (t < 2048) ? x[((size_t)b * 2048 + t) * 1024 + d]
                       : persist[(size_t)(t - 2048) * 1024 + d];
  kv[i] = f2bf(v);
}

__global__ void transpose_w2_k(const float* __restrict__ src, bf16_t* __restrict__ dst) {
  // src (1024 x 64) -> dst (64 x 1024)
  int i = blockIdx.x * 256 + threadIdx.x;
  int m = i / 1024, d = i % 1024;
  dst[i] = f2bf(src[(size_t)d * 64 + m]);
}

__global__ void zero_k(float* p, int n) {
  int i = blockIdx.x * 256 + threadIdx.x;
  if (i < n) p[i] = 0.f;
}

// C[n1,n2] += sum_r A[r,n1]*B[r,n2]  (split-K with f32 atomics)
__global__ __launch_bounds__(256) void atb_k(const bf16_t* __restrict__ A,
                                             const bf16_t* __restrict__ B,
                                             float* __restrict__ C,
                                             int N1, int N2, int rpc) {
  int idx = blockIdx.x * 256 + threadIdx.x;
  int n1 = idx / N2, n2 = idx % N2;
  int r0 = blockIdx.y * rpc;
  float acc = 0.f;
  for (int r = r0; r < r0 + rpc; r++)
    acc += bf2f(A[(size_t)r * N1 + n1]) * bf2f(B[(size_t)r * N2 + n2]);
  atomicAdd(&C[idx], acc);
}

__global__ void colsum_k(const bf16_t* __restrict__ A, float* __restrict__ out,
                         int N, int rpc) {
  int col = blockIdx.x * 256 + threadIdx.x;
  if (col >= N) return;
  int r0 = blockIdx.y * rpc;
  float s = 0.f;
  for (int r = r0; r < r0 + rpc; r++) s += bf2f(A[(size_t)r * N + col]);
  atomicAdd(&out[col], s);
}

__global__ void update_k(const float* __restrict__ p, const float* __restrict__ g,
                         float* of, bf16_t* ob, int n) {
  int i = blockIdx.x * 256 + threadIdx.x;
  if (i >= n) return;
  float gv = fminf(fmaxf(g[i], -1.f), 1.f);
  float v = p[i] - LRATE * (gv + WDECAY * p[i]);
  if (of) of[i] = v;
  if (ob) ob[i] = f2bf(v);
}

__global__ __launch_bounds__(256) void layernorm_k(const float* __restrict__ x,
                                                   const float* __restrict__ g,
                                                   const float* __restrict__ gamma,
                                                   const float* __restrict__ beta,
                                                   float* __restrict__ out) {
  __shared__ float red[256];
  const int row = blockIdx.x, tid = threadIdx.x;
  const float* xr = x + (size_t)row * 1024;
  const float* gr = g + (size_t)row * 1024;
  float hv[4], s = 0.f, sq = 0.f;
#pragma unroll
  for (int i = 0; i < 4; i++) {
    float v = xr[tid + i * 256] + gr[tid + i * 256];
    hv[i] = v; s += v; sq += v * v;
  }
  red[tid] = s; __syncthreads();
  for (int o2 = 128; o2 > 0; o2 >>= 1) { if (tid < o2) red[tid] += red[tid + o2]; __syncthreads(); }
  float mean = red[0] / 1024.f; __syncthreads();
  red[tid] = sq; __syncthreads();
  for (int o2 = 128; o2 > 0; o2 >>= 1) { if (tid < o2) red[tid] += red[tid + o2]; __syncthreads(); }
  float var = red[0] / 1024.f - mean * mean;
  float inv = rsqrtf(var + 1e-5f);
#pragma unroll
  for (int i = 0; i < 4; i++) {
    int c = tid + i * 256;
    out[(size_t)row * 1024 + c] = (hv[i] - mean) * inv * gamma[c] + beta[c];
  }
}

// ---------------------------------------------------------------------------
extern "C" void kernel_launch(void* const* d_in, const int* in_sizes, int n_in,
                              void* d_out, int out_size, void* d_ws, size_t ws_size,
                              hipStream_t stream) {
  (void)in_sizes; (void)n_in; (void)out_size;
  const float* x   = (const float*)d_in[0];
  const float* W1  = (const float*)d_in[1];
  const float* b1  = (const float*)d_in[2];
  const float* W2  = (const float*)d_in[3];
  const float* b2  = (const float*)d_in[4];
  const float* Wq  = (const float*)d_in[5];
  const float* ipw = (const float*)d_in[6];
  const float* ipb = (const float*)d_in[7];
  const float* opw = (const float*)d_in[8];
  const float* opb = (const float*)d_in[9];
  const float* per = (const float*)d_in[10];
  const float* Wg  = (const float*)d_in[11];
  const float* bg  = (const float*)d_in[12];
  const float* gam = (const float*)d_in[13];
  const float* bet = (const float*)d_in[14];

  constexpr int Bb = 4, S = 2048, D = 1024, Mm = 64, H = 8, Pp = 16;
  constexpr int T  = S + Pp;         // 2064
  constexpr int BS = Bb * S;         // 8192
  constexpr int BT = Bb * T;         // 8256

  // bump allocator over d_ws
  char* w = (char*)d_ws;
  auto alloc = [&](size_t bytes) -> void* {
    void* p = (void*)w;
    w += (bytes + 255) & ~(size_t)255;
    return p;
  };
  bf16_t* xb    = (bf16_t*)alloc((size_t)BS * D * 2);
  bf16_t* kvb   = (bf16_t*)alloc((size_t)BT * D * 2);
  bf16_t* W1b   = (bf16_t*)alloc((size_t)Mm * D * 2);
  bf16_t* W2b   = (bf16_t*)alloc((size_t)D * Mm * 2);
  bf16_t* W2tb  = (bf16_t*)alloc((size_t)Mm * D * 2);
  bf16_t* Wqb   = (bf16_t*)alloc((size_t)D * D * 2);
  bf16_t* ipwb  = (bf16_t*)alloc((size_t)3 * D * D * 2);
  bf16_t* opwb  = (bf16_t*)alloc((size_t)D * D * 2);
  bf16_t* Wgb   = (bf16_t*)alloc((size_t)D * D * 2);
  float*  z1    = (float*)alloc((size_t)BS * Mm * 4);
  bf16_t* hsb   = (bf16_t*)alloc((size_t)BS * Mm * 2);   // reused as hq
  bf16_t* dpb   = (bf16_t*)alloc((size_t)BS * D * 2);    // dpred, reused as queries
  bf16_t* dhb   = (bf16_t*)alloc((size_t)BS * Mm * 2);
  float*  gW1   = (float*)alloc((size_t)Mm * D * 4);
  float*  gW2   = (float*)alloc((size_t)D * Mm * 4);
  float*  db1   = (float*)alloc(Mm * 4);
  float*  db2   = (float*)alloc(D * 4);
  bf16_t* W1ub  = (bf16_t*)alloc((size_t)Mm * D * 2);
  bf16_t* W2ub  = (bf16_t*)alloc((size_t)D * Mm * 2);
  float*  b1u   = (float*)alloc(Mm * 4);
  float*  b2u   = (float*)alloc(D * 4);
  float*  memF  = (float*)alloc((size_t)BS * D * 4);     // later reused as `gated`
  bf16_t* qb    = (bf16_t*)alloc((size_t)BS * D * 2);
  bf16_t* kb    = (bf16_t*)alloc((size_t)BT * D * 2);
  bf16_t* vb    = (bf16_t*)alloc((size_t)BT * D * 2);
  bf16_t* ctxb  = (bf16_t*)alloc((size_t)BS * D * 2);
  bf16_t* msb   = (bf16_t*)alloc((size_t)BS * D * 2);
  if ((size_t)(w - (char*)d_ws) > ws_size) return;   // workspace too small

  auto conv = [&](const float* s, bf16_t* d, int n) {
    f32_to_bf16_k<<<(n + 255) / 256, 256, 0, stream>>>(s, d, n);
  };
#define GEMM(EPI, X_, W_, B_, M_, N_, K_, FO_, BO_, AUX_, SCL_)                           \
  gemm_bf16_k<EPI><<<dim3((M_) / 64, (N_) / 64), 256, 0, stream>>>(                       \
      X_, W_, B_, M_, N_, K_, FO_, BO_, AUX_, SCL_)

  // ---- stage inputs to bf16 ----
  conv(x, xb, BS * D);
  build_kv_k<<<(BT * D) / 256, 256, 0, stream>>>(x, per, kvb);
  conv(W1, W1b, Mm * D);
  conv(W2, W2b, D * Mm);
  transpose_w2_k<<<(Mm * D) / 256, 256, 0, stream>>>(W2, W2tb);
  conv(Wq, Wqb, D * D);
  conv(ipw, ipwb, 3 * D * D);
  conv(opw, opwb, D * D);
  conv(Wg, Wgb, D * D);

  // ---- neural memory write: forward + manual backward ----
  GEMM(2, xb, W1b, b1, BS, Mm, D, z1, hsb, (const float*)nullptr, 0.f);        // z1, silu(z1)
  GEMM(3, hsb, W2b, b2, BS, D, Mm, (float*)nullptr, dpb, x, 2.f / 8388608.f);  // dpred (bf16)
  zero_k<<<(D + 255) / 256, 256, 0, stream>>>(db2, D);
  colsum_k<<<dim3((D + 255) / 256, 32), 256, 0, stream>>>(dpb, db2, D, BS / 32);
  GEMM(4, dpb, W2tb, (const float*)nullptr, BS, Mm, D, (float*)nullptr, dhb, z1, 0.f);
  zero_k<<<1, 256, 0, stream>>>(db1, Mm);
  colsum_k<<<dim3(1, 32), 256, 0, stream>>>(dhb, db1, Mm, BS / 32);
  zero_k<<<(Mm * D) / 256, 256, 0, stream>>>(gW1, Mm * D);
  atb_k<<<dim3((Mm * D) / 256, 32), 256, 0, stream>>>(dhb, xb, gW1, Mm, D, BS / 32);
  zero_k<<<(D * Mm) / 256, 256, 0, stream>>>(gW2, D * Mm);
  atb_k<<<dim3((D * Mm) / 256, 32), 256, 0, stream>>>(dpb, hsb, gW2, D, Mm, BS / 32);
  update_k<<<(Mm * D + 255) / 256, 256, 0, stream>>>(W1, gW1, (float*)nullptr, W1ub, Mm * D);
  update_k<<<(D * Mm + 255) / 256, 256, 0, stream>>>(W2, gW2, (float*)nullptr, W2ub, D * Mm);
  update_k<<<1, 256, 0, stream>>>(b1, db1, b1u, (bf16_t*)nullptr, Mm);
  update_k<<<(D + 255) / 256, 256, 0, stream>>>(b2, db2, b2u, (bf16_t*)nullptr, D);

  // ---- memory read ----
  GEMM(1, xb, Wqb, (const float*)nullptr, BS, D, D, (float*)nullptr, dpb, (const float*)nullptr, 0.f); // queries
  GEMM(5, dpb, W1ub, b1u, BS, Mm, D, (float*)nullptr, hsb, (const float*)nullptr, 0.f);                // silu hidden
  GEMM(0, hsb, W2ub, b2u, BS, D, Mm, memF, (bf16_t*)nullptr, (const float*)nullptr, 0.f);              // mem (f32)

  // ---- attention projections ----
  GEMM(1, xb, ipwb, ipb, BS, D, D, (float*)nullptr, qb, (const float*)nullptr, 0.f);
  GEMM(1, kvb, ipwb + (size_t)D * D, ipb + D, BT, D, D, (float*)nullptr, kb, (const float*)nullptr, 0.f);
  GEMM(1, kvb, ipwb + (size_t)2 * D * D, ipb + 2 * D, BT, D, D, (float*)nullptr, vb, (const float*)nullptr, 0.f);

  // ---- flash attention ----
  flash_attn_k<<<dim3(S / 128, Bb * H), 256, 0, stream>>>(qb, kb, vb, ctxb);

  // ---- out_proj (+ mem residual), gate, layernorm ----
  GEMM(6, ctxb, opwb, opb, BS, D, D, memF, msb, (const float*)nullptr, 0.f);           // mem += attn_out
  GEMM(0, msb, Wgb, bg, BS, D, D, memF, (bf16_t*)nullptr, (const float*)nullptr, 0.f); // gated (alias)
  layernorm_k<<<BS, 256, 0, stream>>>(x, memF, gam, bet, (float*)d_out);
#undef GEMM
}
